// Decoder_27006754357970
// MI455X (gfx1250) — compile-verified, matrix operand order checked by reading, not run on hardware
//
#include <hip/hip_runtime.h>
#include <hip/hip_bf16.h>
#include <math.h>

typedef __attribute__((ext_vector_type(16))) __bf16 v16bf;
typedef __attribute__((ext_vector_type(8)))  __bf16 v8bf;
typedef __attribute__((ext_vector_type(8)))  float  v8f;

#define HID   1024
#define VOC   32000
#define TLEN  30
#define BSZ   64
#define H3    (3 * HID)
#define LDSTR 40   // LDS row stride in shorts: 32 data + 8 pad (80B, conflict-free)

__device__ __forceinline__ unsigned short f2bf(float f) {
    unsigned int u = __float_as_uint(f);
    u += 0x7FFFu + ((u >> 16) & 1u);          // round-to-nearest-even
    return (unsigned short)(u >> 16);
}

// CDNA5 async copy: 16B global -> LDS, tracked by ASYNCcnt
__device__ __forceinline__ void async_ld16(unsigned lds_byte_addr,
                                           const unsigned short* g) {
    asm volatile("global_load_async_to_lds_b128 %0, %1, off"
                 :: "v"(lds_byte_addr), "v"(g) : "memory");
}
__device__ __forceinline__ void wait_async0() {
    asm volatile("s_wait_asynccnt 0x0" ::: "memory");
}

__device__ __forceinline__ v16bf ld_lds_frag(const unsigned short* p) {
    v8bf lo = *(const v8bf*)p;          // ds_load_b128
    v8bf hi = *(const v8bf*)(p + 8);    // ds_load_b128
    return __builtin_shufflevector(lo, hi, 0,1,2,3,4,5,6,7,8,9,10,11,12,13,14,15);
}

#define WMMA_BF16(A0, B0, CC) \
    __builtin_amdgcn_wmma_f32_16x16x32_bf16(false, (A0), false, (B0), (short)0, (CC), false, false)

// ---------------- fp32 -> bf16 conversion (weights, h0) ----------------
__global__ void cvt_bf16_kernel(const float* __restrict__ src,
                                unsigned short* __restrict__ dst, long n) {
    long i = (long)blockIdx.x * blockDim.x + threadIdx.x;
    long stride = (long)gridDim.x * blockDim.x;
    for (; i < n; i += stride) dst[i] = f2bf(src[i]);
}

// ---------------- token gather + embedding + relu -> bf16 --------------
__global__ void embed_relu_kernel(const int* __restrict__ target,
                                  const float* __restrict__ emb,
                                  unsigned short* __restrict__ Xb) {
    int row = blockIdx.x;                  // t*B + b
    int t = row / BSZ, b = row % BSZ;
    int tok = (t == 0) ? 0 : target[b * TLEN + (t - 1)];   // SOS_TOKEN = 0
    const float* e = emb + (long)tok * HID;
    unsigned short* o = Xb + (long)row * HID;
    for (int h = threadIdx.x; h < HID; h += blockDim.x) {
        float x = e[h];
        o[h] = f2bf(x > 0.f ? x : 0.f);
    }
}

// ---------------- bf16 WMMA GEMM: C[M,N] = A[M,K] * B[N,K]^T + bias ----
// block = 256 thr (8 waves); block tile M=64 x N=256, K-slab = 32,
// LDS double-buffered via global_load_async_to_lds_b128 (ASYNCcnt).
// mode 0: C row-major [M,N]
// mode 1: logits scatter: row = t*B+b  ->  C[((b*TLEN)+t)*VOC + col]
__global__ void gemm_bf16_kernel(const unsigned short* __restrict__ A,
                                 const unsigned short* __restrict__ Bm,
                                 const float* __restrict__ bias,
                                 float* __restrict__ C,
                                 int M, int N, int K, int mode) {
    __shared__ unsigned short sA[2][64 * LDSTR];    //  2 x  5 KB
    __shared__ unsigned short sB[2][256 * LDSTR];   //  2 x 20 KB

    const int tid  = threadIdx.x;
    const int lane = tid & 31;
    const int w    = tid >> 5;
    const int half = lane >> 4;
    const int lr   = lane & 15;
    const int m0b  = blockIdx.y * 64;               // block row base
    const int n0b  = blockIdx.x * 256;              // block col base
    const int wm   = (w & 1) * 32;                  // wave row offset in block
    const int wn   = (w >> 1) * 64;                 // wave col offset in block

    // per-thread staging coordinates (16B per async load)
    const int srow = tid >> 2;                      // 0..63
    const int seg  = tid & 3;                       // 0..3 (16B segments of 64B row)

    // precomputed fragment offsets (shorts) within one LDS buffer
    const int offA0 = (wm + lr) * LDSTR + half * 16;
    const int offA1 = (wm + 16 + lr) * LDSTR + half * 16;
    const int offB  = (wn + lr) * LDSTR + half * 16;

    v8f acc[2][4];
#pragma unroll
    for (int i = 0; i < 2; ++i)
#pragma unroll
        for (int j = 0; j < 4; ++j) acc[i][j] = (v8f){0.f,0.f,0.f,0.f,0.f,0.f,0.f,0.f};

    const int nslab = K / 32;

    // ---- stage slab 0 into buffer 0 ----
    {
        async_ld16((unsigned)(uintptr_t)&sA[0][srow * LDSTR + seg * 8],
                   A + (size_t)(m0b + srow) * K + seg * 8);
#pragma unroll
        for (int it = 0; it < 4; ++it) {
            int col = srow + it * 64;
            async_ld16((unsigned)(uintptr_t)&sB[0][col * LDSTR + seg * 8],
                       Bm + (size_t)(n0b + col) * K + seg * 8);
        }
    }
    wait_async0();
    __syncthreads();

    for (int s = 0; s < nslab; ++s) {
        const int cur = s & 1, nxt = cur ^ 1;
        // ---- prefetch next slab asynchronously ----
        if (s + 1 < nslab) {
            const int kk = (s + 1) * 32;
            async_ld16((unsigned)(uintptr_t)&sA[nxt][srow * LDSTR + seg * 8],
                       A + (size_t)(m0b + srow) * K + kk + seg * 8);
#pragma unroll
            for (int it = 0; it < 4; ++it) {
                int col = srow + it * 64;
                async_ld16((unsigned)(uintptr_t)&sB[nxt][col * LDSTR + seg * 8],
                           Bm + (size_t)(n0b + col) * K + kk + seg * 8);
            }
        }
        // ---- load ALL fragments first (one dscnt wait), then 8 WMMAs ----
        const unsigned short* sAc = sA[cur];
        const unsigned short* sBc = sB[cur];
        v16bf a0 = ld_lds_frag(sAc + offA0);
        v16bf a1 = ld_lds_frag(sAc + offA1);
        v16bf b0 = ld_lds_frag(sBc + offB);
        v16bf b1 = ld_lds_frag(sBc + offB + 16 * LDSTR);
        v16bf b2 = ld_lds_frag(sBc + offB + 32 * LDSTR);
        v16bf b3 = ld_lds_frag(sBc + offB + 48 * LDSTR);

        acc[0][0] = WMMA_BF16(a0, b0, acc[0][0]);
        acc[1][0] = WMMA_BF16(a1, b0, acc[1][0]);
        acc[0][1] = WMMA_BF16(a0, b1, acc[0][1]);
        acc[1][1] = WMMA_BF16(a1, b1, acc[1][1]);
        acc[0][2] = WMMA_BF16(a0, b2, acc[0][2]);
        acc[1][2] = WMMA_BF16(a1, b2, acc[1][2]);
        acc[0][3] = WMMA_BF16(a0, b3, acc[0][3]);
        acc[1][3] = WMMA_BF16(a1, b3, acc[1][3]);

        // my async writes complete + all waves done reading `cur`
        wait_async0();
        __syncthreads();
    }

    // ---- epilogue: C layout vgpr r / lane half -> row r + 8*half ----
#pragma unroll
    for (int i = 0; i < 2; ++i) {
        int mbase = m0b + wm + i * 16;
#pragma unroll
        for (int j = 0; j < 4; ++j) {
            int col = n0b + wn + j * 16 + lr;
            float bv = bias ? bias[col] : 0.f;
#pragma unroll
            for (int r = 0; r < 8; ++r) {
                int row = mbase + r + half * 8;
                float vq = acc[i][j][r] + bv;
                if (mode == 0) {
                    C[(long)row * N + col] = vq;
                } else {
                    int tt = row >> 6;                // row / 64
                    int bb = row & 63;
                    C[((long)bb * TLEN + tt) * (long)VOC + col] = vq;
                }
            }
        }
    }
}

// ---------------- GRU gate fusion for one timestep ----------------------
__global__ void gru_gate_kernel(const float* __restrict__ Gi,      // [B,3H] (this step)
                                const float* __restrict__ Gh,      // [B,3H]
                                const float* __restrict__ hprev,   // [B,H] fp32
                                float* __restrict__ hout,          // [B,H] fp32
                                unsigned short* __restrict__ hout16) {
    int idx = blockIdx.x * blockDim.x + threadIdx.x;   // B*H
    if (idx >= BSZ * HID) return;
    int b = idx / HID, h = idx % HID;
    const float* gi = Gi + (long)b * H3;
    const float* gh = Gh + (long)b * H3;
    float ir = gi[h],           hr = gh[h];
    float iz = gi[HID + h],     hz = gh[HID + h];
    float in = gi[2 * HID + h], hn = gh[2 * HID + h];
    float r = 1.f / (1.f + expf(-(ir + hr)));
    float z = 1.f / (1.f + expf(-(iz + hz)));
    float n = tanhf(in + r * hn);
    float hv = (1.f - z) * n + z * hprev[idx];
    hout[idx]   = hv;
    hout16[idx] = f2bf(hv);
}

// ---------------- in-place row log-softmax over V ----------------------
__global__ void log_softmax_kernel(float* __restrict__ out) {
    long base = (long)blockIdx.x * VOC;
    __shared__ float red[256];
    float m = -INFINITY;
    for (int i = threadIdx.x; i < VOC; i += 256) m = fmaxf(m, out[base + i]);
    red[threadIdx.x] = m; __syncthreads();
    for (int s = 128; s > 0; s >>= 1) {
        if (threadIdx.x < s) red[threadIdx.x] = fmaxf(red[threadIdx.x], red[threadIdx.x + s]);
        __syncthreads();
    }
    m = red[0]; __syncthreads();
    float sum = 0.f;
    for (int i = threadIdx.x; i < VOC; i += 256) sum += expf(out[base + i] - m);
    red[threadIdx.x] = sum; __syncthreads();
    for (int s = 128; s > 0; s >>= 1) {
        if (threadIdx.x < s) red[threadIdx.x] += red[threadIdx.x + s];
        __syncthreads();
    }
    float lse = m + logf(red[0]);
    for (int i = threadIdx.x; i < VOC; i += 256) out[base + i] -= lse;
}

__global__ void copy_f32_kernel(const float* __restrict__ src,
                                float* __restrict__ dst, int n) {
    int i = blockIdx.x * blockDim.x + threadIdx.x;
    if (i < n) dst[i] = src[i];
}

extern "C" void kernel_launch(void* const* d_in, const int* in_sizes, int n_in,
                              void* d_out, int out_size, void* d_ws, size_t ws_size,
                              hipStream_t stream) {
    const float* enc_hidden = (const float*)d_in[1];   // [1,B,H]
    const int*   target     = (const int*)d_in[2];     // [B,T]
    const float* embedding  = (const float*)d_in[3];   // [V,H]
    const float* w_ih       = (const float*)d_in[4];   // [3H,H]
    const float* w_hh       = (const float*)d_in[5];   // [3H,H]
    const float* b_ih       = (const float*)d_in[6];   // [3H]
    const float* b_hh       = (const float*)d_in[7];   // [3H]
    const float* out_w      = (const float*)d_in[8];   // [V,H]
    const float* out_b      = (const float*)d_in[9];   // [V]
    float* out = (float*)d_out;                        // [B,T,V] + [1,B,H]

    char* ws = (char*)d_ws;
    size_t off = 0;
    auto alloc = [&](size_t bytes) -> void* {
        void* p = ws + off;
        off = (off + bytes + 255) & ~(size_t)255;
        return p;
    };
    unsigned short* wih16  = (unsigned short*)alloc((size_t)H3 * HID * 2);
    unsigned short* whh16  = (unsigned short*)alloc((size_t)H3 * HID * 2);
    unsigned short* outw16 = (unsigned short*)alloc((size_t)VOC * HID * 2);
    unsigned short* Xb     = (unsigned short*)alloc((size_t)TLEN * BSZ * HID * 2);
    unsigned short* Hb     = (unsigned short*)alloc((size_t)(TLEN + 1) * BSZ * HID * 2); // slot 0 = h0
    float*          Hf     = (float*)alloc((size_t)TLEN * BSZ * HID * 4);
    float*          Gi     = (float*)alloc((size_t)TLEN * BSZ * H3 * 4);
    float*          Gh     = (float*)alloc((size_t)BSZ * H3 * 4);

    // 1. convert weights + h0 to bf16
    cvt_bf16_kernel<<<2048, 256, 0, stream>>>(w_ih, wih16, (long)H3 * HID);
    cvt_bf16_kernel<<<2048, 256, 0, stream>>>(w_hh, whh16, (long)H3 * HID);
    cvt_bf16_kernel<<<4096, 256, 0, stream>>>(out_w, outw16, (long)VOC * HID);
    cvt_bf16_kernel<<<256, 256, 0, stream>>>(enc_hidden, Hb, (long)BSZ * HID);

    // 2. embed + relu for all timesteps
    embed_relu_kernel<<<TLEN * BSZ, 256, 0, stream>>>(target, embedding, Xb);

    // 3. Gi = X @ w_ih^T + b_ih for all steps at once  (M=1920, N=3072, K=1024)
    gemm_bf16_kernel<<<dim3(H3 / 256, TLEN * BSZ / 64), 256, 0, stream>>>(
        Xb, wih16, b_ih, Gi, TLEN * BSZ, H3, HID, 0);

    // 4. sequential GRU recurrence
    for (int t = 0; t < TLEN; ++t) {
        const unsigned short* hprev16 = Hb + (size_t)t * BSZ * HID;
        gemm_bf16_kernel<<<dim3(H3 / 256, 1), 256, 0, stream>>>(
            hprev16, whh16, b_hh, Gh, BSZ, H3, HID, 0);
        const float* hprevf = (t == 0) ? enc_hidden : (Hf + (size_t)(t - 1) * BSZ * HID);
        gru_gate_kernel<<<(BSZ * HID) / 256, 256, 0, stream>>>(
            Gi + (size_t)t * BSZ * H3, Gh, hprevf,
            Hf + (size_t)t * BSZ * HID,
            Hb + (size_t)(t + 1) * BSZ * HID);
    }

    // 5. logits for all steps: [1920,1024] x [1024,32000], scatter to [B,T,V] + out_b
    gemm_bf16_kernel<<<dim3(VOC / 256, TLEN * BSZ / 64), 256, 0, stream>>>(
        Hb + (size_t)BSZ * HID, outw16, out_b, out, TLEN * BSZ, VOC, HID, 1);

    // 6. in-place log-softmax per (b,t) row
    log_softmax_kernel<<<BSZ * TLEN, 256, 0, stream>>>(out);

    // 7. final hidden state appended after log_probs
    copy_f32_kernel<<<(BSZ * HID) / 256, 256, 0, stream>>>(
        Hf + (size_t)(TLEN - 1) * BSZ * HID,
        out + (size_t)BSZ * TLEN * VOC, BSZ * HID);
}
